// GCN_47579647705688
// MI455X (gfx1250) — matrix operand order repaired and checked by reading, not compiled, not run on hardware
//
#include <hip/hip_runtime.h>
#include <hip/hip_bf16.h>
#include <math.h>

typedef __attribute__((ext_vector_type(16))) _Float16 v16h;
typedef __attribute__((ext_vector_type(8)))  _Float16 v8h;
typedef __attribute__((ext_vector_type(8)))  float    v8f;

#define F0   10    // input features
#define F1   128   // conv channels
#define FC   138   // concat width (10 + 128)
#define FCP  160   // concat width padded to multiple of 32 (WMMA K-step)
#define FH   1024  // MLP hidden

// ---------------------------------------------------------------------------
// WMMA A-tile loader. 16-bit A 16x32 in this code's contiguous k order:
//   lane = (m) + 16*?? : lanes 0-15 hold M=0..15, halfs j map to
//   K = 16*hi + j, hi = lane>>4 (matches B's contiguous order).
// So each lane needs two contiguous 8-half (16 B) chunks from row m:
//   [kbase + 16*hi, +8) and [kbase + 16*hi + 8, +8).
// ---------------------------------------------------------------------------
__device__ __forceinline__ v16h wmma_load_a(const _Float16* __restrict__ A,
                                            int lda, int row, int kbase, int hi) {
  const _Float16* p = A + (size_t)row * lda + kbase + 16 * hi;
  union { v16h v; v8h h[2]; } u;
  u.h[0] = *(const v8h*)(p);        // K = kbase + 16*hi + 0..7
  u.h[1] = *(const v8h*)(p + 8);    // K = kbase + 16*hi + 8..15
  return u.v;
}

// ---------------------------------------------------------------------------
// Degree / normalization
// ---------------------------------------------------------------------------
__global__ void k_deg_init(float* __restrict__ deg, int n) {
  int i = blockIdx.x * blockDim.x + threadIdx.x;
  if (i < n) deg[i] = 1.0f;  // self-loop
}

__global__ void k_deg_count(const int* __restrict__ dst, float* __restrict__ deg, int E) {
  int e = blockIdx.x * blockDim.x + threadIdx.x;
  if (e < E) atomicAdd(&deg[dst[e]], 1.0f);
}

__global__ void k_dinv(const float* __restrict__ deg, float* __restrict__ dinv, int n) {
  int i = blockIdx.x * blockDim.x + threadIdx.x;
  if (i < n) dinv[i] = rsqrtf(deg[i]);
}

// ---------------------------------------------------------------------------
// GEMM1: t = x @ W1   (K=10, tiny -> VALU with W1 staged in LDS)
// one thread per output element; 128 threads share one row of x.
// ---------------------------------------------------------------------------
__global__ void k_gemm1(const float* __restrict__ x, const float* __restrict__ W1,
                        float* __restrict__ t, int n) {
  __shared__ float sW[F0 * F1];
  for (int i = threadIdx.x; i < F0 * F1; i += blockDim.x) sW[i] = W1[i];
  __syncthreads();
  int g = blockIdx.x * blockDim.x + threadIdx.x;
  int i = g >> 7, c = g & 127;
  if (i >= n) return;
  const float* xr = x + (size_t)i * F0;
  float acc = 0.f;
#pragma unroll
  for (int k = 0; k < F0; ++k) acc = fmaf(xr[k], sW[k * F1 + c], acc);
  t[(size_t)i * F1 + c] = acc;
}

// ---------------------------------------------------------------------------
// Aggregation: agg[i] = t[i]*dinv[i]^2 (self loop), then edges add
// t[s]*dinv[s]*dinv[d] into agg[d].  float4 per thread, 4 f32 atomics.
// ---------------------------------------------------------------------------
__global__ void k_self_init(const float* __restrict__ t, const float* __restrict__ dinv,
                            float* __restrict__ agg, int n) {
  long long g = (long long)blockIdx.x * blockDim.x + threadIdx.x;
  int i = (int)(g >> 5);
  if (i >= n) return;
  int c = ((int)g & 31) * 4;
  float w = dinv[i]; w *= w;
  float4 v = *(const float4*)(t + (size_t)i * F1 + c);
  v.x *= w; v.y *= w; v.z *= w; v.w *= w;
  *(float4*)(agg + (size_t)i * F1 + c) = v;
}

__global__ void k_edge_agg(const float* __restrict__ t, const int* __restrict__ src,
                           const int* __restrict__ dst, const float* __restrict__ dinv,
                           float* __restrict__ agg, int E) {
  long long g = (long long)blockIdx.x * blockDim.x + threadIdx.x;
  int e = (int)(g >> 5);
  if (e >= E) return;
  int c = ((int)g & 31) * 4;
  int s = src[e], d = dst[e];
  float nrm = dinv[s] * dinv[d];
  float4 v = *(const float4*)(t + (size_t)s * F1 + c);
  float* ap = agg + (size_t)d * F1 + c;
  atomicAdd(ap + 0, v.x * nrm);
  atomicAdd(ap + 1, v.y * nrm);
  atomicAdd(ap + 2, v.z * nrm);
  atomicAdd(ap + 3, v.w * nrm);
}

// finish layer 1: h1 = f16(relu(agg + b1))
__global__ void k_finish1(const float* __restrict__ agg, const float* __restrict__ b,
                          _Float16* __restrict__ h1h, int n) {
  long long g = (long long)blockIdx.x * blockDim.x + threadIdx.x;
  int i = (int)(g >> 7);
  if (i >= n) return;
  int c = (int)g & 127;
  float v = agg[(size_t)i * F1 + c] + b[c];
  h1h[(size_t)i * F1 + c] = (_Float16)(v > 0.f ? v : 0.f);
}

// finish layer 2: build padded concat  hc[i][0..9]=x, [10..137]=relu(agg+b2), [138..159]=0
__global__ void k_finish2(const float* __restrict__ agg, const float* __restrict__ b,
                          const float* __restrict__ x, _Float16* __restrict__ hch, int n) {
  long long g = (long long)blockIdx.x * blockDim.x + threadIdx.x;
  long long total = (long long)n * FCP;
  if (g >= total) return;
  int i = (int)(g / FCP);
  int c = (int)(g - (long long)i * FCP);
  float v;
  if (c < F0) {
    v = x[(size_t)i * F0 + c];
  } else if (c < FC) {
    v = agg[(size_t)i * F1 + (c - F0)] + b[c - F0];
    v = v > 0.f ? v : 0.f;
  } else {
    v = 0.f;
  }
  hch[(size_t)i * FCP + c] = (_Float16)v;
}

// ---------------------------------------------------------------------------
// Weight prep: transposed f16 copies, [N][K] so each lane's 16 K-contiguous
// halfs for the WMMA B layout are one 32-byte load.
// ---------------------------------------------------------------------------
__global__ void k_prep_w2(const float* __restrict__ W2, _Float16* __restrict__ W2t) {
  int g = blockIdx.x * blockDim.x + threadIdx.x;   // 128*128
  int nn = g >> 7, k = g & 127;
  if (nn >= F1) return;
  W2t[(size_t)nn * F1 + k] = (_Float16)W2[(size_t)k * F1 + nn];
}

__global__ void k_prep_wl1(const float* __restrict__ WL1, _Float16* __restrict__ WL1t) {
  int g = blockIdx.x * blockDim.x + threadIdx.x;   // 1024*160
  int nn = g / FCP, k = g - nn * FCP;
  if (nn >= FH) return;
  float v = (k < FC) ? WL1[(size_t)k * FH + nn] : 0.f;
  WL1t[(size_t)nn * FCP + k] = (_Float16)v;
}

// ---------------------------------------------------------------------------
// GEMM2: t2[N][128] = h1f16[N][128] @ W2  via v_wmma_f32_16x16x32_f16
// one 16x16 output tile per wave, K=128 -> 4 WMMA ops.
// rows16 = N/16 (N=50000 divides exactly; no partial tiles -> EXEC all ones)
// ---------------------------------------------------------------------------
__global__ void k_gemm2_wmma(const _Float16* __restrict__ A, const _Float16* __restrict__ Bt,
                             float* __restrict__ C, int rows16) {
  int gw = (blockIdx.x * blockDim.x + threadIdx.x) >> 5;
  int lane = threadIdx.x & 31;
  int tm = gw >> 3;       // 8 column tiles (128/16)
  int tn = gw & 7;
  if (tm >= rows16) return;
  int m = lane & 15, hi = lane >> 4;
  int row0 = tm * 16;
  v8f acc = {};
#pragma unroll
  for (int ks = 0; ks < 4; ++ks) {
    v16h a = wmma_load_a(A, F1, row0 + m, ks * 32, hi);
    // B 32x16 layout: lane = n + 16*(k>>4), half j = k&15  -> Bt[n][kbase+16*hi .. +16)
    v16h b = *(const v16h*)(Bt + (size_t)(tn * 16 + m) * F1 + ks * 32 + 16 * hi);
    acc = __builtin_amdgcn_wmma_f32_16x16x32_f16(false, a, false, b, (short)0, acc,
                                                 false, false);
  }
  // C/D layout: VGPR r -> row r+8*hi, lane&15 -> col
  float* cp = C + (size_t)row0 * F1 + tn * 16 + m;
#pragma unroll
  for (int r = 0; r < 8; ++r)
    cp[(size_t)(r + 8 * hi) * F1] = acc[r];
}

// ---------------------------------------------------------------------------
// Fused MLP: out = sigmoid( relu(hc @ WL1 + bL1) @ WL2 + bL2 )
// Block = 16 rows, 8 waves; wave w covers column tiles {w, w+8, ..., w+56}.
// Hidden activations (N x 1024) never leave registers.
// ---------------------------------------------------------------------------
__global__ void k_mlp_fused(const _Float16* __restrict__ A,   // [N][160] f16
                            const _Float16* __restrict__ Bt,  // [1024][160] f16
                            const float* __restrict__ bL1,    // [1024]
                            const float* __restrict__ wL2,    // [1024]
                            const float* __restrict__ bL2,    // [1]
                            float* __restrict__ out, int n) {
  __shared__ float srow[16];
  int lane = threadIdx.x & 31;
  int wave = threadIdx.x >> 5;     // 0..7
  int row0 = blockIdx.x * 16;
  if (threadIdx.x < 16) srow[threadIdx.x] = 0.f;
  __syncthreads();

  int m = lane & 15, hi = lane >> 4;
  // A tile (16 rows x 160 K) in registers, reused across all 8 column tiles
  v16h a[5];
#pragma unroll
  for (int ks = 0; ks < 5; ++ks) a[ks] = wmma_load_a(A, FCP, row0 + m, ks * 32, hi);

  float accrow[8];
#pragma unroll
  for (int r = 0; r < 8; ++r) accrow[r] = 0.f;

  for (int tIdx = 0; tIdx < 8; ++tIdx) {
    int n0 = (wave + tIdx * 8) * 16;
    v8f acc = {};
#pragma unroll
    for (int ks = 0; ks < 5; ++ks) {
      v16h b = *(const v16h*)(Bt + (size_t)(n0 + m) * FCP + ks * 32 + 16 * hi);
      acc = __builtin_amdgcn_wmma_f32_16x16x32_f16(false, a[ks], false, b, (short)0, acc,
                                                   false, false);
    }
    int col = n0 + m;                 // this lane's output column
    float bias = bL1[col];
    float w2 = wL2[col];
#pragma unroll
    for (int r = 0; r < 8; ++r) {
      float v = acc[r] + bias;        // hidden unit for row r+8*hi, col
      v = v > 0.f ? v : 0.f;          // relu
      accrow[r] = fmaf(v, w2, accrow[r]);
    }
  }
  // reduce over the 16 lanes sharing `hi` (xor over lane bits 0..3 keeps hi fixed)
#pragma unroll
  for (int off = 1; off < 16; off <<= 1)
#pragma unroll
    for (int r = 0; r < 8; ++r)
      accrow[r] += __shfl_xor(accrow[r], off, 32);

  if (m == 0) {
#pragma unroll
    for (int r = 0; r < 8; ++r)
      atomicAdd(&srow[r + 8 * hi], accrow[r]);   // ds_add_f32
  }
  __syncthreads();

  if (threadIdx.x < 16) {
    int row = row0 + threadIdx.x;
    if (row < n) {
      float s = srow[threadIdx.x] + bL2[0];
      out[row] = 1.0f / (1.0f + expf(-s));
    }
  }
}

// ---------------------------------------------------------------------------
extern "C" void kernel_launch(void* const* d_in, const int* in_sizes, int n_in,
                              void* d_out, int out_size, void* d_ws, size_t ws_size,
                              hipStream_t stream) {
  const float* x   = (const float*)d_in[0];
  const float* W1  = (const float*)d_in[1];
  const float* b1  = (const float*)d_in[2];
  const float* W2  = (const float*)d_in[3];
  const float* b2  = (const float*)d_in[4];
  const float* WL1 = (const float*)d_in[5];
  const float* bL1 = (const float*)d_in[6];
  const float* WL2 = (const float*)d_in[7];
  const float* bL2 = (const float*)d_in[8];
  const int*   ei  = (const int*)d_in[9];

  const int N = in_sizes[0] / F0;
  const int E = in_sizes[9] / 2;
  const int* src = ei;
  const int* dst = ei + E;

  // workspace carve-up (256B aligned)
  char* ws = (char*)d_ws;
  size_t off = 0;
  auto carve = [&](size_t bytes) -> void* {
    void* p = ws + off;
    off += (bytes + 255) & ~(size_t)255;
    return p;
  };
  float*    deg  = (float*)carve((size_t)N * 4);
  float*    dinv = (float*)carve((size_t)N * 4);
  float*    tbuf = (float*)carve((size_t)N * F1 * 4);   // GEMM outputs (both layers)
  float*    agg  = (float*)carve((size_t)N * F1 * 4);   // aggregation (both layers)
  _Float16* h1h  = (_Float16*)carve((size_t)N * F1 * 2);
  _Float16* hch  = (_Float16*)carve((size_t)N * FCP * 2);
  _Float16* W2t  = (_Float16*)carve((size_t)F1 * F1 * 2);
  _Float16* WL1t = (_Float16*)carve((size_t)FH * FCP * 2);

  const int B = 256;
  auto nblk = [](long long work, int blk) { return (int)((work + blk - 1) / blk); };

  // --- normalization ---
  k_deg_init <<<nblk(N, B), B, 0, stream>>>(deg, N);
  k_deg_count<<<nblk(E, B), B, 0, stream>>>(dst, deg, E);
  k_dinv     <<<nblk(N, B), B, 0, stream>>>(deg, dinv, N);

  // --- weight prep (f16 transposed) ---
  k_prep_w2 <<<nblk((long long)F1 * F1, B), B, 0, stream>>>(W2, W2t);
  k_prep_wl1<<<nblk((long long)FH * FCP, B), B, 0, stream>>>(WL1, WL1t);

  // --- GCN layer 1 ---
  k_gemm1    <<<nblk((long long)N * F1, B), B, 0, stream>>>(x, W1, tbuf, N);
  k_self_init<<<nblk((long long)N * 32, B), B, 0, stream>>>(tbuf, dinv, agg, N);
  k_edge_agg <<<nblk((long long)E * 32, B), B, 0, stream>>>(tbuf, src, dst, dinv, agg, E);
  k_finish1  <<<nblk((long long)N * F1, B), B, 0, stream>>>(agg, b1, h1h, N);

  // --- GCN layer 2 (WMMA GEMM) ---
  int rows16 = N / 16;                      // N=50000 -> 3125, exact
  k_gemm2_wmma<<<nblk((long long)rows16 * 8 * 32, B), B, 0, stream>>>(h1h, W2t, tbuf, rows16);
  k_self_init <<<nblk((long long)N * 32, B), B, 0, stream>>>(tbuf, dinv, agg, N);
  k_edge_agg  <<<nblk((long long)E * 32, B), B, 0, stream>>>(tbuf, src, dst, dinv, agg, E);
  k_finish2   <<<nblk((long long)N * FCP, B), B, 0, stream>>>(agg, b2, x, hch, N);

  // --- fused MLP head (WMMA) ---
  k_mlp_fused<<<rows16, B, 0, stream>>>(hch, WL1t, bL1, WL2, bL2, (float*)d_out, N);
}
